// SSMMemoryLLM_56109452755121
// MI455X (gfx1250) — matrix operand order repaired and checked by reading, not compiled, not run on hardware
//
#include <hip/hip_runtime.h>
#include <hip/hip_fp16.h>

// ---------------- problem constants (from reference) ----------------
#define DM   1024          // d_model
#define DI   2048          // d_inner
#define DS   16            // d_state
#define DC   4             // d_conv
#define BB   2             // batch
#define LL   2048          // seq
#define MM   (BB*LL)       // 4096 rows
#define NC   64            // scan chunks
#define CL   (LL/NC)       // 32 steps per chunk

typedef _Float16 h16 __attribute__((ext_vector_type(16)));
typedef _Float16 h8  __attribute__((ext_vector_type(8)));
typedef float    f8  __attribute__((ext_vector_type(8)));

// ---------------- CDNA5 async-copy helpers ----------------
// LDS byte offset of a __shared__ object (addrspacecast generic -> AS3)
__device__ __forceinline__ unsigned lds_off(const void* p) {
    return (unsigned)(size_t)(const __attribute__((address_space(3))) void*)p;
}
// GLOBAL_LOAD_ASYNC_TO_LDS_B128: 16B memory -> LDS, tracked by ASYNCcnt
__device__ __forceinline__ void async_ld_b128(unsigned dst_lds, const void* gptr) {
    asm volatile("global_load_async_to_lds_b128 %0, %1, off"
                 :: "v"(dst_lds), "v"((unsigned long long)(size_t)gptr)
                 : "memory");
}
__device__ __forceinline__ void wait_asynccnt0() {
    asm volatile("s_wait_asynccnt 0x0" ::: "memory");
}

// ---------------- fp32 -> fp16 conversion (weights + activations) ----------------
__global__ __launch_bounds__(256) void cvt_f32_f16(const float* __restrict__ s,
                                                   _Float16* __restrict__ d, int n) {
    int i = blockIdx.x * blockDim.x + threadIdx.x;
    int stride = gridDim.x * blockDim.x;
    for (; i < n; i += stride) d[i] = (_Float16)s[i];
}

// ---------------- WMMA GEMM: C[M,N] = A[M,K] * W[N,K]^T (f16 in, f32 out) ------
// Block = 4 waves = 128x64 output tile; each wave owns 32 rows x 64 cols.
// The 64x32 f16 B K-slice (4KB) is staged per block into LDS via async copies,
// double-buffered across K steps; waves build B fragments with ds_load_b128.
// Fragment layouts per CDNA5 ISA:
//  A 16x32 f16: lane l -> row m=l%16; elems 0..7 = K (l/16)*8..+7, elems 8..15 = +16
//  B 32x16 f16: lane l -> col n=l%16; elems 0..7 = K (l/16)*8..+7, elems 8..15 = +16  (B[k][n]=W[n][k])
//  C/D 16x16 f32: vgpr r, lane l -> row r + 8*(l/16), col l%16
__global__ __launch_bounds__(128) void gemm_wmma_f16(
    const _Float16* __restrict__ A, int lda,
    const _Float16* __restrict__ W, int ldb,
    float* __restrict__ C, int ldc,
    int Mdim, int Ndim, int Kdim)
{
    __shared__ _Float16 sB[2][64][32];     // [buf][W row within N tile][K slice]

    const int tid  = threadIdx.x;
    const int lane = tid & 31;
    const int wv   = tid >> 5;             // wave 0..3 -> M sub-tile
    const int tilesN = Ndim >> 6;
    const int m0 = (blockIdx.x / tilesN) * 128 + wv * 32;
    const int n0 = (blockIdx.x % tilesN) * 64;
    const int kh  = lane >> 4;             // half-wave selector
    const int l16 = lane & 15;

    f8 acc[2][4];
    #pragma unroll
    for (int mi = 0; mi < 2; ++mi)
        #pragma unroll
        for (int ni = 0; ni < 4; ++ni)
            #pragma unroll
            for (int r = 0; r < 8; ++r) acc[mi][ni][r] = 0.0f;

    // prologue: stage K-slice 0 into buffer 0 (256 x 16B chunks, 2 per thread)
    #pragma unroll
    for (int q = tid; q < 256; q += 128) {
        int r = q >> 2, part = q & 3;
        async_ld_b128(lds_off(&sB[0][r][part*8]),
                      W + (size_t)(n0 + r) * ldb + part*8);
    }

    for (int k0 = 0; k0 < Kdim; k0 += 32) {
        const int cur = (k0 >> 5) & 1;
        wait_asynccnt0();                  // our chunks for sB[cur] have landed
        __syncthreads();                   // everyone's landed; sB[cur^1] is free
        if (k0 + 32 < Kdim) {              // stage next K-slice into other buffer
            #pragma unroll
            for (int q = tid; q < 256; q += 128) {
                int r = q >> 2, part = q & 3;
                async_ld_b128(lds_off(&sB[cur ^ 1][r][part*8]),
                              W + (size_t)(n0 + r) * ldb + (k0 + 32) + part*8);
            }
        }

        h16 a[2];
        #pragma unroll
        for (int mi = 0; mi < 2; ++mi) {
            const _Float16* ap = A + (size_t)(m0 + mi*16 + l16) * lda + k0 + kh*8;
            __builtin_prefetch(ap + 32, 0, 1);    // global_prefetch next K step
            h8 lo = *(const h8*)(ap);
            h8 hi = *(const h8*)(ap + 16);
            #pragma unroll
            for (int e = 0; e < 8; ++e) { a[mi][e] = lo[e]; a[mi][8+e] = hi[e]; }
        }
        h16 bm[4];
        #pragma unroll
        for (int ni = 0; ni < 4; ++ni) {
            const _Float16* bp = &sB[cur][ni*16 + l16][kh*8];    // ds_load_b128 x2
            h8 lo = *(const h8*)(bp);
            h8 hi = *(const h8*)(bp + 16);
            #pragma unroll
            for (int e = 0; e < 8; ++e) { bm[ni][e] = lo[e]; bm[ni][8+e] = hi[e]; }
        }
        #pragma unroll
        for (int mi = 0; mi < 2; ++mi)
            #pragma unroll
            for (int ni = 0; ni < 4; ++ni)
                acc[mi][ni] = __builtin_amdgcn_wmma_f32_16x16x32_f16(
                    false, a[mi], false, bm[ni], (short)0, acc[mi][ni], false, false);
    }

    #pragma unroll
    for (int mi = 0; mi < 2; ++mi)
        #pragma unroll
        for (int ni = 0; ni < 4; ++ni)
            #pragma unroll
            for (int r = 0; r < 8; ++r) {
                int row = m0 + mi*16 + r + kh*8;
                int col = n0 + ni*16 + l16;
                C[(size_t)row * ldc + col] = acc[mi][ni][r];
            }
}

// ---------------- causal depthwise conv(4) + bias + SiLU ----------------
__global__ __launch_bounds__(256) void conv_silu(
    const float* __restrict__ xz, const float* __restrict__ cw,
    const float* __restrict__ cb, float* __restrict__ xc)
{
    int idx = blockIdx.x * blockDim.x + threadIdx.x;
    if (idx >= MM * DI) return;
    int d = idx % DI;
    int m = idx / DI;
    int l = m % LL;
    float acc = cb[d];
    #pragma unroll
    for (int j = 0; j < DC; ++j) {
        int ll = l - (DC - 1) + j;
        if (ll >= 0)
            acc += cw[d*DC + j] * xz[(size_t)(m - (DC - 1) + j) * (2*DI) + d];
    }
    float sg = 1.0f / (1.0f + __expf(-acc));
    xc[(size_t)m * DI + d] = acc * sg;
}

// ---------------- x_dbl[m, 0..32] = xc[m,:] . W_x[e,:]  (wave per output) ------
__global__ __launch_bounds__(256) void xdbl_gemv(
    const float* __restrict__ xc, const float* __restrict__ Wx,
    float* __restrict__ xdbl)
{
    const int lane = threadIdx.x & 31;
    const int wave = blockIdx.x * (blockDim.x >> 5) + (threadIdx.x >> 5);
    if (wave >= MM * 33) return;
    int m = wave / 33, e = wave % 33;
    const float* xr = xc + (size_t)m * DI;
    const float* wr = Wx + (size_t)e * DI;
    float s = 0.0f;
    for (int k = lane; k < DI; k += 32) s += xr[k] * wr[k];
    #pragma unroll
    for (int off = 16; off; off >>= 1) s += __shfl_xor(s, off, 32);
    if (lane == 0) xdbl[(size_t)m * 33 + e] = s;
}

// ---------------- chunked scan ----------------
// lane layout per block: 256 threads = 16 channels x 16 states (n fastest).
__device__ __forceinline__ float softplus_f(float x) {
    return (x > 20.0f) ? x : __logf(1.0f + __expf(x));
}

// pass 1: per chunk c compute P = prod a_t  and S = local scan with h_in = 0
__global__ __launch_bounds__(256) void scan_pass1(
    const float* __restrict__ xc, const float* __restrict__ xdbl,
    const float* __restrict__ Wdt, const float* __restrict__ bdt,
    const float* __restrict__ Alog,
    float* __restrict__ Pb, float* __restrict__ Sb)
{
    __shared__ float sBC[33];
    const int tid  = threadIdx.x;
    const int n    = tid & 15;
    const int dloc = tid >> 4;
    const int c    = blockIdx.x % NC;
    const int grp  = blockIdx.x / NC;
    const int b    = grp / (DI/16);
    const int d    = (grp % (DI/16)) * 16 + dloc;
    const float An  = -__expf(Alog[d*DS + n]);
    const float wdt = Wdt[d], bd = bdt[d];
    float h = 0.0f, p = 1.0f;
    for (int tt = 0; tt < CL; ++tt) {
        const int t = c*CL + tt;
        const size_t row = (size_t)(b*LL + t);
        if (tid < 33) sBC[tid] = xdbl[row*33 + tid];
        __syncthreads();
        float dt = softplus_f(sBC[0]*wdt + bd);
        float Bn = sBC[1 + n];
        float xt = xc[row*DI + d];
        float a  = __expf(dt * An);
        h = a*h + dt*Bn*xt;
        p = p * a;
        __syncthreads();
    }
    const size_t idx = ((size_t)(b*DI + d)*DS + n)*NC + c;
    Pb[idx] = p; Sb[idx] = h;
}

// pass 2: 64-step prefix per (b,d,n): h_in[c] for each chunk
__global__ __launch_bounds__(256) void scan_pass2(
    const float* __restrict__ Pb, const float* __restrict__ Sb,
    float* __restrict__ Hin)
{
    int idx = blockIdx.x * blockDim.x + threadIdx.x;   // (b*DI+d)*DS + n
    if (idx >= BB*DI*DS) return;
    const size_t base = (size_t)idx * NC;
    float h = 0.0f;
    for (int c = 0; c < NC; ++c) {
        Hin[base + c] = h;
        h = Pb[base + c] * h + Sb[base + c];
    }
}

// pass 3: replay chunk with true incoming state; y = h.C, +D*x, *silu(z) -> f16
__global__ __launch_bounds__(256) void scan_pass3(
    const float* __restrict__ xc, const float* __restrict__ xz,
    const float* __restrict__ xdbl, const float* __restrict__ Wdt,
    const float* __restrict__ bdt, const float* __restrict__ Alog,
    const float* __restrict__ Dp, const float* __restrict__ Hin,
    _Float16* __restrict__ y16)
{
    __shared__ float sBC[33];
    const int tid  = threadIdx.x;
    const int n    = tid & 15;
    const int dloc = tid >> 4;
    const int c    = blockIdx.x % NC;
    const int grp  = blockIdx.x / NC;
    const int b    = grp / (DI/16);
    const int d    = (grp % (DI/16)) * 16 + dloc;
    const float An  = -__expf(Alog[d*DS + n]);
    const float wdt = Wdt[d], bd = bdt[d], Dv = Dp[d];
    float h = Hin[((size_t)(b*DI + d)*DS + n)*NC + c];
    for (int tt = 0; tt < CL; ++tt) {
        const int t = c*CL + tt;
        const size_t row = (size_t)(b*LL + t);
        if (tid < 33) sBC[tid] = xdbl[row*33 + tid];
        __syncthreads();
        float dt = softplus_f(sBC[0]*wdt + bd);
        float Bn = sBC[1 + n];
        float Cn = sBC[17 + n];
        float xt = xc[row*DI + d];
        h = __expf(dt*An)*h + dt*Bn*xt;
        float y = h * Cn;
        #pragma unroll
        for (int off = 8; off; off >>= 1) y += __shfl_xor(y, off, 16);
        if (n == 0) {
            float zv = xz[row*(2*DI) + DI + d];
            float out = (y + Dv*xt) * (zv / (1.0f + __expf(-zv)));
            y16[row*DI + d] = (_Float16)out;
        }
        __syncthreads();
    }
}

// ---------------- launcher ----------------
extern "C" void kernel_launch(void* const* d_in, const int* in_sizes, int n_in,
                              void* d_out, int out_size, void* d_ws, size_t ws_size,
                              hipStream_t stream) {
    const float* x      = (const float*)d_in[0];   // (2,2048,1024)
    const float* W_in   = (const float*)d_in[1];   // (4096,1024)
    const float* conv_w = (const float*)d_in[2];   // (2048,1,4)
    const float* conv_b = (const float*)d_in[3];   // (2048,)
    const float* W_x    = (const float*)d_in[4];   // (33,2048)
    const float* W_dt   = (const float*)d_in[5];   // (2048,1)
    const float* b_dt   = (const float*)d_in[6];   // (2048,)
    const float* A_log  = (const float*)d_in[7];   // (2048,16)
    const float* Dp     = (const float*)d_in[8];   // (2048,)
    const float* W_out  = (const float*)d_in[9];   // (1024,2048)
    float* out = (float*)d_out;                    // (2,2048,1024) f32

    // workspace carve-up (256B aligned slices)
    size_t off = 0;
    auto take = [&](size_t bytes) {
        char* p = (char*)d_ws + off;
        off += (bytes + 255) & ~(size_t)255;
        return (void*)p;
    };
    _Float16* xh    = (_Float16*)take((size_t)MM*DM*2);
    _Float16* Winh  = (_Float16*)take((size_t)(2*DI)*DM*2);
    _Float16* Wouth = (_Float16*)take((size_t)DM*DI*2);
    float*    xzf   = (float*)  take((size_t)MM*(2*DI)*4);
    float*    xcf   = (float*)  take((size_t)MM*DI*4);
    float*    xdb   = (float*)  take((size_t)MM*33*4);
    float*    Pb    = (float*)  take((size_t)BB*DI*DS*NC*4);
    float*    Sb    = (float*)  take((size_t)BB*DI*DS*NC*4);
    float*    Hin   = (float*)  take((size_t)BB*DI*DS*NC*4);
    _Float16* y16   = (_Float16*)take((size_t)MM*DI*2);
    (void)ws_size; (void)in_sizes; (void)n_in; (void)out_size;

    // 1) fp16 copies of GEMM operands
    cvt_f32_f16<<<4096, 256, 0, stream>>>(x,     xh,    MM*DM);
    cvt_f32_f16<<<4096, 256, 0, stream>>>(W_in,  Winh,  (2*DI)*DM);
    cvt_f32_f16<<<4096, 256, 0, stream>>>(W_out, Wouth, DM*DI);

    // 2) xz = x @ W_in^T : M=4096, N=4096, K=1024 -> (4096/128)*(4096/64) blocks
    gemm_wmma_f16<<<(MM/128)*((2*DI)/64), 128, 0, stream>>>(
        xh, DM, Winh, DM, xzf, 2*DI, MM, 2*DI, DM);

    // 3) depthwise conv + SiLU
    conv_silu<<<(MM*DI + 255)/256, 256, 0, stream>>>(xzf, conv_w, conv_b, xcf);

    // 4) x_dbl = xi @ W_x^T (skinny): one wave per (m,e)
    xdbl_gemv<<<(MM*33 + 7)/8, 256, 0, stream>>>(xcf, W_x, xdb);

    // 5) chunked selective scan
    scan_pass1<<<(BB*DI/16)*NC, 256, 0, stream>>>(xcf, xdb, W_dt, b_dt, A_log, Pb, Sb);
    scan_pass2<<<(BB*DI*DS + 255)/256, 256, 0, stream>>>(Pb, Sb, Hin);
    scan_pass3<<<(BB*DI/16)*NC, 256, 0, stream>>>(xcf, xzf, xdb, W_dt, b_dt, A_log,
                                                  Dp, Hin, y16);

    // 6) out = y @ W_out^T : M=4096, N=1024, K=2048 -> (4096/128)*(1024/64) blocks
    gemm_wmma_f16<<<(MM/128)*(DM/64), 128, 0, stream>>>(
        y16, DI, Wouth, DI, out, DM, MM, DM, DI);
}
